// MultiHeadAttention_58437325030125
// MI455X (gfx1250) — matrix operand order repaired and checked by reading, not compiled, not run on hardware
//
#include <hip/hip_runtime.h>

#define S_LEN 1024
#define HIDN  1024
#define NHEAD 16
#define DKH   64
#define BSZ   4
#define MTOK  (BSZ * S_LEN)  // 4096

typedef __attribute__((ext_vector_type(16))) __bf16 v16bf;
typedef __attribute__((ext_vector_type(8)))  __bf16 v8bf;
typedef __attribute__((ext_vector_type(8)))  float  v8f;

__device__ __forceinline__ v8f wmma_bf16(v16bf a, v16bf b, v8f c) {
  // D = A(16x32) * B(32x16) + C, f32 accumulate
  return __builtin_amdgcn_wmma_f32_16x16x32_bf16(false, a, false, b, (short)0, c, false, false);
}

// ---- CDNA5 async global->LDS copy (ASYNCcnt-tracked), 16B per lane ----
__device__ __forceinline__ void async_ld_b128(void* lds, const void* g) {
  unsigned int l = (unsigned int)(unsigned long long)lds;  // low 32b = LDS offset
  asm volatile("global_load_async_to_lds_b128 %0, %1, off"
               :: "v"(l), "v"(g) : "memory");
}
__device__ __forceinline__ void wait_async0() {
  asm volatile("s_wait_asynccnt 0x0" ::: "memory");
}
__device__ __forceinline__ void wait_async1() {
  asm volatile("s_wait_asynccnt 0x1" ::: "memory");
}
__device__ __forceinline__ void wait_async4() {
  asm volatile("s_wait_asynccnt 0x4" ::: "memory");
}

// ---- WMMA fragment loaders (contiguous k = (lane>>4)*16 + e in both A and B) ----
// A (16x32 bf16): lane holds row m = lane&15; with base = k0 + (lane>>4)*16,
// elems 0..15 = K[base..base+15].
__device__ __forceinline__ v16bf frag_a(const __bf16* row, int base) {
  v8bf lo = *(const v8bf*)(row + base);
  v8bf hi = *(const v8bf*)(row + base + 8);
  v16bf r;
#pragma unroll
  for (int i = 0; i < 8; ++i) { r[i] = lo[i]; r[i + 8] = hi[i]; }
  return r;
}
// B (32x16 bf16): lane holds column n = lane&15, K = (lane>>4)*16 + e ->
// 16 contiguous K-values from the source pointer.
__device__ __forceinline__ v16bf frag_b(const __bf16* p) {
  return *(const v16bf*)p;
}

// ---------------------------------------------------------------------------
// One-shot f32 -> bf16 conversion (batch + weights), removes converts from
// all GEMM/attention hot loops and halves their read bytes.
// ---------------------------------------------------------------------------
__global__ __launch_bounds__(256) void cvt_f32_bf16_k(
    const float* __restrict__ src, __bf16* __restrict__ dst, int n)
{
  int i = (blockIdx.x * 256 + threadIdx.x) * 8;
  if (i < n) {
#pragma unroll
    for (int j = 0; j < 8; ++j) dst[i + j] = (__bf16)src[i + j];
  }
}

// ---------------------------------------------------------------------------
// C[M,N] = A[M,K] * W[N,K]^T + bias[N]   (M=4096, N=K=1024, all bf16 inputs)
// Block = 8 waves = 128(M) x 64(N) tile.  B tile (64x32 bf16, 4KB) is staged
// in LDS via async-to-LDS copies, double-buffered, shared by all 8 waves.
// MODE 0: f32 out [M,N]   MODE 1: bf16 [b,h,s,d]   MODE 2: bf16 [b,h,d,s]
// ---------------------------------------------------------------------------
template <int MODE>
__global__ __launch_bounds__(256) void gemm_bf16_k(
    const __bf16* __restrict__ A, const __bf16* __restrict__ W,
    const float* __restrict__ bias, float* __restrict__ outF,
    __bf16* __restrict__ outH)
{
  __shared__ __align__(32) __bf16 Bs[2][64 * 32];  // 2 x 4KB double buffer
  const int K = HIDN, N = HIDN;
  const int tid  = threadIdx.x;
  const int wv   = tid >> 5;
  const int lane = tid & 31;
  const int lr   = lane & 15, hf = lane >> 4;
  const int bm = blockIdx.x / (N / 64);
  const int bn = blockIdx.x % (N / 64);
  const int m0 = bm * 128 + wv * 16;
  const int n0 = bn * 64;

  // one 16B async chunk per thread per stage: 64 rows x 4 chunks
  const int srow = tid >> 2;
  const int soff = (tid & 3) * 8;
  const __bf16* wsrc = W + (size_t)(n0 + srow) * K + soff;

  async_ld_b128(&Bs[0][srow * 32 + soff], wsrc);  // stage k-tile 0

  const __bf16* arow = A + (size_t)(m0 + lr) * K;
  v8f acc[4] = {};
  const int NK = K / 32;
  for (int ks = 0; ks < NK; ++ks) {
    if (ks + 1 < NK) {  // prefetch next tile into other buffer
      async_ld_b128(&Bs[(ks + 1) & 1][srow * 32 + soff], wsrc + (ks + 1) * 32);
      wait_async1();    // in-order: retires the current tile's copy
    } else {
      wait_async0();
    }
    __syncthreads();
    const __bf16* bs = &Bs[ks & 1][0];
    const v16bf av = frag_a(arow, ks * 32 + hf * 16);
    acc[0] = wmma_bf16(av, frag_b(bs + ( 0 + lr) * 32 + hf * 16), acc[0]);
    acc[1] = wmma_bf16(av, frag_b(bs + (16 + lr) * 32 + hf * 16), acc[1]);
    acc[2] = wmma_bf16(av, frag_b(bs + (32 + lr) * 32 + hf * 16), acc[2]);
    acc[3] = wmma_bf16(av, frag_b(bs + (48 + lr) * 32 + hf * 16), acc[3]);
    __syncthreads();
  }
#pragma unroll
  for (int nt = 0; nt < 4; ++nt) {
    const int n = n0 + nt * 16 + lr;
    const float bv = bias[n];
#pragma unroll
    for (int r = 0; r < 8; ++r) {
      const int m = m0 + r + 8 * hf;           // C/D layout: M = r + 8*(lane>>4)
      const float v = acc[nt][r] + bv;
      if constexpr (MODE == 0) {
        outF[(size_t)m * N + n] = v;
      } else {
        const int b = m >> 10, s = m & (S_LEN - 1);
        const int h = n >> 6,  d = n & (DKH - 1);
        if constexpr (MODE == 1)
          outH[(((size_t)b * NHEAD + h) * S_LEN + s) * DKH + d] = (__bf16)v;
        else
          outH[(((size_t)b * NHEAD + h) * DKH + d) * S_LEN + s] = (__bf16)v;
      }
    }
  }
}

// ---------------------------------------------------------------------------
// Flash-style attention.  Block = 4 waves on one (b,h), 64 query rows.
// Per 32-key step, K tile (32x64) and V tile (64x32) are staged in LDS via
// async copies (double-buffered) and shared by all 4 waves; attn_bias is
// streamed from HBM exactly once (the roofline term) with prefetch.
// ---------------------------------------------------------------------------
__global__ __launch_bounds__(128) void attn_fwd_k(
    const __bf16* __restrict__ Qh, const __bf16* __restrict__ Kh,
    const __bf16* __restrict__ Vt, const float* __restrict__ bias,
    __bf16* __restrict__ Ob)
{
  __shared__ __align__(32) __bf16 Kt[2][32 * 64];   // 2 x 4KB
  __shared__ __align__(32) __bf16 Vts[2][64 * 32];  // 2 x 4KB
  __shared__ __align__(32) __bf16 pbuf[4][16 * 32]; // per-wave P staging
  const int tid  = threadIdx.x;
  const int wv   = tid >> 5;
  const int lane = tid & 31;
  const int lr = lane & 15, hf = lane >> 4;
  const int nqb = S_LEN / 64;
  const int bh = blockIdx.x / nqb;
  const int qb = blockIdx.x % nqb;
  const int q0 = qb * 64 + wv * 16;

  const __bf16* Qb = Qh + (size_t)bh * S_LEN * DKH;
  const __bf16* Kb = Kh + (size_t)bh * S_LEN * DKH;
  const __bf16* Vb = Vt + (size_t)bh * DKH * S_LEN;
  const float*  Bb = bias + (size_t)bh * S_LEN * S_LEN;

  // staging: 256 chunks of 16B per tile, 128 threads -> 2 chunks each, x2 tiles
  const int c0 = tid * 2, c1 = tid * 2 + 1;
  const int kr0 = c0 >> 3, ko0 = (c0 & 7) * 8, kr1 = c1 >> 3, ko1 = (c1 & 7) * 8;
  const int vr0 = c0 >> 2, vo0 = (c0 & 3) * 8, vr1 = c1 >> 2, vo1 = (c1 & 3) * 8;

  // stage tiles for kb = 0
  async_ld_b128(&Kt[0][kr0 * 64 + ko0], Kb + (size_t)kr0 * DKH + ko0);
  async_ld_b128(&Kt[0][kr1 * 64 + ko1], Kb + (size_t)kr1 * DKH + ko1);
  async_ld_b128(&Vts[0][vr0 * 32 + vo0], Vb + (size_t)vr0 * S_LEN + vo0);
  async_ld_b128(&Vts[0][vr1 * 32 + vo1], Vb + (size_t)vr1 * S_LEN + vo1);

  const __bf16* qrow = Qb + (size_t)(q0 + lr) * DKH;
  const v16bf aq0 = frag_a(qrow, hf * 16);         // d = 0..31
  const v16bf aq1 = frag_a(qrow, 32 + hf * 16);    // d = 32..63

  v8f acc[4] = {};                                // O accumulator 16x64
  float rowM[8], rowL[8];
#pragma unroll
  for (int r = 0; r < 8; ++r) { rowM[r] = -3.0e38f; rowL[r] = 0.f; }

  __bf16* pb = &pbuf[wv][0];
  const int NKB = S_LEN / 32;
  for (int ks = 0; ks < NKB; ++ks) {
    const int kb = ks * 32;
    if (ks + 1 < NKB) {  // prefetch next K/V tiles
      const int nb = (ks + 1) & 1, kn = kb + 32;
      async_ld_b128(&Kt[nb][kr0 * 64 + ko0], Kb + (size_t)(kn + kr0) * DKH + ko0);
      async_ld_b128(&Kt[nb][kr1 * 64 + ko1], Kb + (size_t)(kn + kr1) * DKH + ko1);
      async_ld_b128(&Vts[nb][vr0 * 32 + vo0], Vb + (size_t)vr0 * S_LEN + kn + vo0);
      async_ld_b128(&Vts[nb][vr1 * 32 + vo1], Vb + (size_t)vr1 * S_LEN + kn + vo1);
      wait_async4();     // in-order: retires this step's 4 copies
      __builtin_prefetch(Bb + (size_t)(q0 + lr) * S_LEN + kb + 32 + hf * 16, 0, 0);
    } else {
      wait_async0();
    }
    __syncthreads();
    const __bf16* kt = &Kt[ks & 1][0];
    const __bf16* vt = &Vts[ks & 1][0];

    // ---- scores: two 16x16 tiles over 32 keys (K-dim = DK, 2 wmma each) ----
    v8f sc0 = {}, sc1 = {};
    sc0 = wmma_bf16(aq0, frag_b(kt + ( 0 + lr) * 64 + hf * 16), sc0);
    sc0 = wmma_bf16(aq1, frag_b(kt + ( 0 + lr) * 64 + 32 + hf * 16), sc0);
    sc1 = wmma_bf16(aq0, frag_b(kt + (16 + lr) * 64 + hf * 16), sc1);
    sc1 = wmma_bf16(aq1, frag_b(kt + (16 + lr) * 64 + 32 + hf * 16), sc1);

    // ---- scale, zero-mask, +bias, online softmax (16-lane reductions) ----
#pragma unroll
    for (int r = 0; r < 8; ++r) {
      const int m = q0 + r + 8 * hf;
      const float* brow = Bb + (size_t)m * S_LEN + kb;
      float x0 = sc0[r] * 0.125f;                 // SCALE = DK^-0.5
      float x1 = sc1[r] * 0.125f;
      if (x0 == 0.f) x0 = -9.0e15f;
      if (x1 == 0.f) x1 = -9.0e15f;
      x0 += brow[lr];
      x1 += brow[16 + lr];
      float t = fmaxf(x0, x1);
#pragma unroll
      for (int i = 8; i >= 1; i >>= 1) t = fmaxf(t, __shfl_xor(t, i, 32));
      const float nm   = fmaxf(rowM[r], t);
      const float corr = __expf(rowM[r] - nm);
      rowM[r] = nm;
      const float p0 = __expf(x0 - nm);
      const float p1 = __expf(x1 - nm);
      float ps = p0 + p1;
#pragma unroll
      for (int i = 8; i >= 1; i >>= 1) ps += __shfl_xor(ps, i, 32);
      rowL[r] = rowL[r] * corr + ps;
      acc[0][r] *= corr; acc[1][r] *= corr; acc[2][r] *= corr; acc[3][r] *= corr;
      pb[(r + 8 * hf) * 32 + lr]      = (__bf16)p0;  // stage P 16x32 row-major
      pb[(r + 8 * hf) * 32 + 16 + lr] = (__bf16)p1;
    }
    __syncthreads();

    // ---- O += P(16x32) * V(32x64): 4 wmma along d ----
    const v16bf ap = frag_a(pb + lr * 32, hf * 16);
    acc[0] = wmma_bf16(ap, frag_b(vt + ( 0 + lr) * 32 + hf * 16), acc[0]);
    acc[1] = wmma_bf16(ap, frag_b(vt + (16 + lr) * 32 + hf * 16), acc[1]);
    acc[2] = wmma_bf16(ap, frag_b(vt + (32 + lr) * 32 + hf * 16), acc[2]);
    acc[3] = wmma_bf16(ap, frag_b(vt + (48 + lr) * 32 + hf * 16), acc[3]);
    __syncthreads();  // all reads of buf[ks&1] done before it is re-staged
  }

  // ---- normalize and emit O as bf16 [token, HID] for the out-projection ----
  const int b = bh / NHEAD, h = bh % NHEAD;
#pragma unroll
  for (int r = 0; r < 8; ++r) {
    const float inv = 1.0f / rowL[r];
    const size_t tok = (size_t)b * S_LEN + q0 + r + 8 * hf;
#pragma unroll
    for (int nt = 0; nt < 4; ++nt) {
      Ob[tok * HIDN + h * DKH + nt * 16 + lr] = (__bf16)(acc[nt][r] * inv);
    }
  }
}

extern "C" void kernel_launch(void* const* d_in, const int* in_sizes, int n_in,
                              void* d_out, int out_size, void* d_ws, size_t ws_size,
                              hipStream_t stream) {
  (void)in_sizes; (void)n_in; (void)out_size; (void)ws_size;
  const float* batch     = (const float*)d_in[0];
  const float* attn_bias = (const float*)d_in[1];
  const float* Wq = (const float*)d_in[2];
  const float* bq = (const float*)d_in[3];
  const float* Wk = (const float*)d_in[4];
  const float* bk = (const float*)d_in[5];
  const float* Wv = (const float*)d_in[6];
  const float* bv = (const float*)d_in[7];
  const float* Wo = (const float*)d_in[8];
  const float* bo = (const float*)d_in[9];
  float* out = (float*)d_out;

  const size_t NE  = (size_t)MTOK * HIDN;  // 4096*1024
  const size_t WSZ = (size_t)HIDN * HIDN;  // 1024*1024
  __bf16* Xh  = (__bf16*)d_ws;             // 8 MB
  __bf16* Qh  = Xh + NE;                   // 8 MB
  __bf16* Kh  = Qh + NE;                   // 8 MB
  __bf16* Vth = Kh + NE;                   // 8 MB
  __bf16* Ob  = Vth + NE;                  // 8 MB
  __bf16* Wqh = Ob + NE;                   // 2 MB x4
  __bf16* Wkh = Wqh + WSZ;
  __bf16* Wvh = Wkh + WSZ;
  __bf16* Woh = Wvh + WSZ;

  // one-shot f32 -> bf16 conversions (L2-resident, tiny vs 268MB bias stream)
  cvt_f32_bf16_k<<<dim3((unsigned)(NE / 2048)), dim3(256), 0, stream>>>(batch, Xh, (int)NE);
  cvt_f32_bf16_k<<<dim3((unsigned)(WSZ / 2048)), dim3(256), 0, stream>>>(Wq, Wqh, (int)WSZ);
  cvt_f32_bf16_k<<<dim3((unsigned)(WSZ / 2048)), dim3(256), 0, stream>>>(Wk, Wkh, (int)WSZ);
  cvt_f32_bf16_k<<<dim3((unsigned)(WSZ / 2048)), dim3(256), 0, stream>>>(Wv, Wvh, (int)WSZ);
  cvt_f32_bf16_k<<<dim3((unsigned)(WSZ / 2048)), dim3(256), 0, stream>>>(Wo, Woh, (int)WSZ);

  const dim3 gg((MTOK / 128) * (HIDN / 64));  // 512 blocks
  const dim3 gb(256);
  gemm_bf16_k<1><<<gg, gb, 0, stream>>>(Xh, Wqh, bq, nullptr, Qh);
  gemm_bf16_k<1><<<gg, gb, 0, stream>>>(Xh, Wkh, bk, nullptr, Kh);
  gemm_bf16_k<2><<<gg, gb, 0, stream>>>(Xh, Wvh, bv, nullptr, Vth);
  attn_fwd_k<<<dim3(BSZ * NHEAD * (S_LEN / 64)), dim3(128), 0, stream>>>(
      Qh, Kh, Vth, attn_bias, Ob);
  gemm_bf16_k<0><<<gg, gb, 0, stream>>>(Ob, Woh, bo, out, nullptr);
}
